// FFC_20624432955796
// MI455X (gfx1250) — matrix operand order repaired and checked by reading, not compiled, not run on hardware
//
#include <hip/hip_runtime.h>
#include <hip/hip_bf16.h>
#include <stdint.h>

#define BQ      1024
#define QN      65536
#define DK      512
#define NCHUNKS 32
#define CHUNK   2048
#define ROWS    128
#define SUBW    64
#define KSTEP   64
#define MARGIN  0.4f
#define SCALE   32.0f
#define HN      10
#define BSTRIDE 72    // shorts; 144B row stride, 16B-multiple
#define TSTRIDE 68    // floats; 272B row stride, 16B-multiple
#define PSTRIDE 22    // floats per (row,chunk) partial: T1,T2,top10_1,top10_2

typedef __attribute__((ext_vector_type(16))) __bf16 v16bf;
typedef __attribute__((ext_vector_type(8)))  float  v8f;

union FragBf { v16bf v; uint4 u[2]; };

// pack two floats to bf16 pair (round-to-nearest-even), low = x, high = y
__device__ __forceinline__ unsigned packbf(float x, float y) {
  unsigned ux = __float_as_uint(x), uy = __float_as_uint(y);
  unsigned lo = (ux + 0x7FFFu + ((ux >> 16) & 1u)) >> 16;
  unsigned hi = (uy + 0x7FFFu + ((uy >> 16) & 1u)) & 0xFFFF0000u;
  return lo | hi;
}

__device__ __forceinline__ void ins10(float* t, float v) {
#pragma unroll
  for (int j = 0; j < HN; ++j) { float hi = fmaxf(t[j], v); v = fminf(t[j], v); t[j] = hi; }
}

// top-10 of union of two sorted-descending 10-lists
__device__ __forceinline__ void mergeTop(float* a, const float* b) {
  float c[HN];
#pragma unroll
  for (int k = 0; k < HN; ++k) {
    float m = fmaxf(a[k], b[k]);
#pragma unroll
    for (int i = 0; i < k; ++i) m = fmaxf(m, fminf(a[i], b[k - 1 - i]));
    c[k] = m;
  }
#pragma unroll
  for (int k = 0; k < HN; ++k) a[k] = c[k];
}

__global__ void ffc_zero_acc(float* acc) {
  if (threadIdx.x < 8) acc[threadIdx.x] = 0.f;
}

// convert p (1024x512 f32) to bf16 once
__global__ __launch_bounds__(256) void ffc_cvt_p(const float* __restrict__ p,
                                                 unsigned short* __restrict__ pbf) {
  const int idx = blockIdx.x * 256 + threadIdx.x;     // 131072 float4s
  float4 f = ((const float4*)p)[idx];
  uint2 pk;
  pk.x = packbf(f.x, f.y);
  pk.y = packbf(f.z, f.w);
  ((uint2*)pbf)[idx] = pk;
}

// ---------------- Phase 1: fused dual-GEMM + fixed-max softmax sums + top-10 ----------------
__global__ __launch_bounds__(256) void ffc_gemm_stats(
    const unsigned short* __restrict__ pbf, const float* __restrict__ queue,
    const float* __restrict__ mask, float* __restrict__ part)
{
  __shared__ unsigned short Bs[2 * SUBW * BSTRIDE];  // queue0/queue1 64-K slice, bf16 (18.4 KB)
  __shared__ float tile[ROWS * TSTRIDE];             // cos tile, one matrix at a time (34.8 KB)
  __shared__ float mbuf[SUBW];

  const int tid  = threadIdx.x;
  const int wave = tid >> 5, lane = tid & 31;
  const int rowTile = blockIdx.x;          // 0..7
  const int chunk   = blockIdx.y;          // 0..31
  const int chunkBase = chunk * CHUNK;

  // A fragment base: wave owns rows wave*16..+15; contiguous k = 16*(lane>>4) + s (matches B)
  const unsigned short* arowp = pbf
      + (size_t)(rowTile * ROWS + wave * 16 + (lane & 15)) * DK + ((lane >> 4) * 16);

  const float* q0 = queue;
  const float* q1 = queue + (size_t)QN * DK;

  // per-thread epilogue ownership: row er (0..127), 32-col slice
  const int er = tid >> 1;
  const int ec = (tid & 1) * 32;

  float T1 = 0.f, T2 = 0.f;
  float tl1[HN], tl2[HN];
#pragma unroll
  for (int k = 0; k < HN; ++k) { tl1[k] = -1e30f; tl2[k] = -1e30f; }

  const int brow = lane & 15;
  const int boff = (lane >> 4) * 16;

  for (int sub = 0; sub < CHUNK / SUBW; ++sub) {
    const int c0g = chunkBase + sub * SUBW;
    v8f S0[4], S1[4];
#pragma unroll
    for (int ct = 0; ct < 4; ++ct) {
      S0[ct] = (v8f){0.f,0.f,0.f,0.f,0.f,0.f,0.f,0.f};
      S1[ct] = (v8f){0.f,0.f,0.f,0.f,0.f,0.f,0.f,0.f};
    }

    for (int kb = 0; kb < DK; kb += KSTEP) {
      __syncthreads();
      { // stage both queue slices: 2 mats x 64 cols x 64 K, f32 -> bf16 RNE
        const int m    = tid >> 7;
        const int rr   = (tid >> 1) & 63;
        const int half = tid & 1;
        const float* src = (m ? q1 : q0) + (size_t)(c0g + rr) * DK + kb + half * 32;
        unsigned short* dst = &Bs[(m * SUBW + rr) * BSTRIDE + half * 32];
#pragma unroll
        for (int i = 0; i < 8; ++i) {
          float4 f = *(const float4*)(src + i * 4);
          uint2 pk;
          pk.x = packbf(f.x, f.y);
          pk.y = packbf(f.z, f.w);
          *(uint2*)(dst + i * 4) = pk;
        }
        if (kb == 0 && tid < SUBW) mbuf[tid] = mask[c0g + tid];
        if (kb + KSTEP < DK) __builtin_prefetch(src + KSTEP, 0, 1);  // global_prefetch_b8
      }
      __syncthreads();

#pragma unroll
      for (int ks = 0; ks < KSTEP; ks += 32) {
        FragBf a;
        a.u[0] = *(const uint4*)(arowp + kb + ks);
        a.u[1] = *(const uint4*)(arowp + kb + ks + 8);
        // hoist all 8 B fragments so one dscnt wait covers them, then WMMAs go back-to-back
        FragBf b0[4], b1[4];
#pragma unroll
        for (int ct = 0; ct < 4; ++ct) {
          const int r0 = (0 * SUBW + ct * 16 + brow) * BSTRIDE + ks + boff;
          const int r1 = (1 * SUBW + ct * 16 + brow) * BSTRIDE + ks + boff;
          b0[ct].u[0] = *(const uint4*)&Bs[r0];
          b0[ct].u[1] = *(const uint4*)&Bs[r0 + 8];
          b1[ct].u[0] = *(const uint4*)&Bs[r1];
          b1[ct].u[1] = *(const uint4*)&Bs[r1 + 8];
        }
#pragma unroll
        for (int ct = 0; ct < 4; ++ct) {
          S0[ct] = __builtin_amdgcn_wmma_f32_16x16x32_bf16(false, a.v, false, b0[ct].v, (short)0, S0[ct], false, false);
          S1[ct] = __builtin_amdgcn_wmma_f32_16x16x32_bf16(false, a.v, false, b1[ct].v, (short)0, S1[ct], false, false);
        }
      }
    }

    // ---- pass 1: cos1 ----
    { // dump (C layout: vgpr i -> row i, +8 for upper lanes; col = lane&15)
      const int rw = wave * 16 + (lane >> 4) * 8;
#pragma unroll
      for (int ct = 0; ct < 4; ++ct) {
        const int c = ct * 16 + (lane & 15);
#pragma unroll
        for (int i = 0; i < 8; ++i) tile[(rw + i) * TSTRIDE + c] = S0[ct][i];
      }
    }
    __syncthreads();
#pragma unroll
    for (int j4 = 0; j4 < 8; ++j4) {
      float4 v = *(const float4*)&tile[er * TSTRIDE + ec + j4 * 4];
      T1 += __expf(__fmaf_rn(v.x, SCALE, -SCALE));
      T1 += __expf(__fmaf_rn(v.y, SCALE, -SCALE));
      T1 += __expf(__fmaf_rn(v.z, SCALE, -SCALE));
      T1 += __expf(__fmaf_rn(v.w, SCALE, -SCALE));
      if (v.x > tl1[HN-1]) ins10(tl1, v.x);
      if (v.y > tl1[HN-1]) ins10(tl1, v.y);
      if (v.z > tl1[HN-1]) ins10(tl1, v.z);
      if (v.w > tl1[HN-1]) ins10(tl1, v.w);
    }
    __syncthreads();

    // ---- pass 2: cos2 = mask*S1 + (1-mask)*S0 ----
    {
      const int rw = wave * 16 + (lane >> 4) * 8;
#pragma unroll
      for (int ct = 0; ct < 4; ++ct) {
        const int c = ct * 16 + (lane & 15);
        const float mk = mbuf[c];
#pragma unroll
        for (int i = 0; i < 8; ++i)
          tile[(rw + i) * TSTRIDE + c] = mk * S1[ct][i] + (1.f - mk) * S0[ct][i];
      }
    }
    __syncthreads();
#pragma unroll
    for (int j4 = 0; j4 < 8; ++j4) {
      float4 v = *(const float4*)&tile[er * TSTRIDE + ec + j4 * 4];
      T2 += __expf(__fmaf_rn(v.x, SCALE, -SCALE));
      T2 += __expf(__fmaf_rn(v.y, SCALE, -SCALE));
      T2 += __expf(__fmaf_rn(v.z, SCALE, -SCALE));
      T2 += __expf(__fmaf_rn(v.w, SCALE, -SCALE));
      if (v.x > tl2[HN-1]) ins10(tl2, v.x);
      if (v.y > tl2[HN-1]) ins10(tl2, v.y);
      if (v.z > tl2[HN-1]) ins10(tl2, v.z);
      if (v.w > tl2[HN-1]) ins10(tl2, v.w);
    }
    // next sub's k-loop barriers protect tile reuse
  }

  // merge the two col-slices of each row (partner = lane^1, same wave)
  T1 += __shfl_xor(T1, 1, 32);
  T2 += __shfl_xor(T2, 1, 32);
  {
    float o[HN];
#pragma unroll
    for (int k = 0; k < HN; ++k) o[k] = __shfl_xor(tl1[k], 1, 32);
    mergeTop(tl1, o);
#pragma unroll
    for (int k = 0; k < HN; ++k) o[k] = __shfl_xor(tl2[k], 1, 32);
    mergeTop(tl2, o);
  }
  if ((tid & 1) == 0) {
    const int rglob = rowTile * ROWS + er;
    float* pr = part + ((size_t)rglob * NCHUNKS + chunk) * PSTRIDE;
    pr[0] = T1; pr[1] = T2;
#pragma unroll
    for (int k = 0; k < HN; ++k) { pr[2 + k] = tl1[k]; pr[12 + k] = tl2[k]; }
  }
}

// ---------------- Phase 2: per-row finalize (exact fp32 GT dot + chunk merge) ----------------
__global__ __launch_bounds__(32) void ffc_finalize_rows(
    const float* __restrict__ p, const float* __restrict__ queue,
    const float* __restrict__ mask, const int* __restrict__ label,
    const float* __restrict__ part, float* __restrict__ acc)
{
  const int b = blockIdx.x;
  const int lane = threadIdx.x;
  const int lbl = label[b];
  const int sl  = (lbl < 0) ? 0 : lbl;

  const float* pb = p + (size_t)b * DK;
  const float* g0 = queue + (size_t)sl * DK;
  const float* g1 = queue + (size_t)QN * DK + (size_t)sl * DK;
  float d0 = 0.f, d1 = 0.f;
#pragma unroll
  for (int j = 0; j < 16; ++j) {
    const float pv = pb[lane * 16 + j];
    d0 = fmaf(pv, g0[lane * 16 + j], d0);
    d1 = fmaf(pv, g1[lane * 16 + j], d1);
  }
#pragma unroll
  for (int s = 1; s <= 16; s <<= 1) { d0 += __shfl_xor(d0, s, 32); d1 += __shfl_xor(d1, s, 32); }
  const float mkl = mask[sl];
  const float gt1 = d0;
  const float gt2 = mkl * d1 + (1.f - mkl) * d0;

  // one chunk partial per lane (NCHUNKS == 32)
  const float* pr = part + ((size_t)b * NCHUNKS + lane) * PSTRIDE;
  float T1 = pr[0], T2 = pr[1];
  float t1[HN], t2[HN];
#pragma unroll
  for (int k = 0; k < HN; ++k) { t1[k] = pr[2 + k]; t2[k] = pr[12 + k]; }

#pragma unroll
  for (int s = 1; s <= 16; s <<= 1) {
    T1 += __shfl_xor(T1, s, 32);
    T2 += __shfl_xor(T2, s, 32);
    float o[HN];
#pragma unroll
    for (int k = 0; k < HN; ++k) o[k] = __shfl_xor(t1[k], s, 32);
    mergeTop(t1, o);
#pragma unroll
    for (int k = 0; k < HN; ++k) o[k] = __shfl_xor(t2[k], s, 32);
    mergeTop(t2, o);
  }

  if (lane == 0) {
    if (lbl >= 0) {
      // replace the un-margined gt term with the margined one, then logsumexp
      const float c1 = T1 - expf(gt1 * SCALE - SCALE) + expf((gt1 - MARGIN) * SCALE - SCALE);
      const float c2 = T2 - expf(gt2 * SCALE - SCALE) + expf((gt2 - MARGIN) * SCALE - SCALE);
      const float L1 = SCALE + logf(c1);
      const float L2 = SCALE + logf(c2);
      atomicAdd(acc + 0, L1 - (gt1 - MARGIN) * SCALE);
      atomicAdd(acc + 1, L2 - (gt2 - MARGIN) * SCALE);
      atomicAdd(acc + 4, 1.f);
    } else {
      float n1 = 0.f, n2 = 0.f;
#pragma unroll
      for (int k = 0; k < HN; ++k) { n1 += fmaxf(t1[k], 0.f); n2 += fmaxf(t2[k], 0.f); }
      atomicAdd(acc + 2, n1 * 0.1f);
      atomicAdd(acc + 3, n2 * 0.1f);
      atomicAdd(acc + 5, 1.f);
    }
  }
}

__global__ void ffc_final_loss(const float* __restrict__ acc, float* __restrict__ out) {
  float loss = 0.f;
  if (acc[4] > 0.f) loss += (acc[0] + acc[1]) / acc[4];
  if (acc[5] > 0.f) loss += (acc[2] + acc[3]) / acc[5];
  out[0] = loss;
}

extern "C" void kernel_launch(void* const* d_in, const int* in_sizes, int n_in,
                              void* d_out, int out_size, void* d_ws, size_t ws_size,
                              hipStream_t stream) {
  (void)in_sizes; (void)n_in; (void)out_size; (void)ws_size;
  const float* p     = (const float*)d_in[0];
  const float* queue = (const float*)d_in[1];
  const float* mask  = (const float*)d_in[2];
  const int*   label = (const int*)d_in[3];

  float* wsf  = (float*)d_ws;
  float* acc  = wsf;                                   // 8 floats
  float* part = wsf + 8;                               // 1024*32*22 floats (~2.9 MB)
  unsigned short* pbf = (unsigned short*)(wsf + 8 + (size_t)BQ * NCHUNKS * PSTRIDE);  // 1 MB

  ffc_zero_acc<<<1, 32, 0, stream>>>(acc);
  ffc_cvt_p<<<(BQ * DK / 4) / 256, 256, 0, stream>>>(p, pbf);
  ffc_gemm_stats<<<dim3(BQ / ROWS, NCHUNKS), 256, 0, stream>>>(pbf, queue, mask, part);
  ffc_finalize_rows<<<BQ, 32, 0, stream>>>(p, queue, mask, label, part, acc);
  ffc_final_loss<<<1, 1, 0, stream>>>(acc, (float*)d_out);
}
